// SDCFlowModelV2_51934744543299
// MI455X (gfx1250) — matrix operand order repaired and checked by reading, not compiled, not run on hardware
//
#include <hip/hip_runtime.h>
#include <hip/hip_bf16.h>

#define B_ 4
#define T_ 32
#define N_ 128
#define F_ 16
#define D_ 64
#define GS_ 128
#define DD_ 64
#define DI_ 32
#define DE_ 32
#define L_ 5
#define NH_ 4

typedef __attribute__((ext_vector_type(16))) _Float16 v16h;
typedef __attribute__((ext_vector_type(8)))  float    v8f;

__device__ __forceinline__ float sigmoidf_(float x) { return 1.0f / (1.0f + __expf(-x)); }
__device__ __forceinline__ float geluf_(float x)    { return 0.5f * x * (1.0f + erff(x * 0.70710678118654752f)); }

// ---------------------------------------------------------------- GRU stack
__global__ __launch_bounds__(64)
void gru_kernel(const float* __restrict__ x,
                const float* __restrict__ g1_Wih, const float* __restrict__ g1_bih,
                const float* __restrict__ g1_Whh, const float* __restrict__ g1_bhh,
                const float* __restrict__ g2_Wih, const float* __restrict__ g2_bih,
                const float* __restrict__ g2_Whh, const float* __restrict__ g2_bhh,
                float* __restrict__ h_local)
{
  int s = blockIdx.x;          // b*N + n
  int b = s >> 7, n = s & 127;
  int d = threadIdx.x;
  __shared__ float xs[F_], h1[D_], h2[D_], s1[D_];
  h1[d] = 0.f; h2[d] = 0.f;
  __syncthreads();
  for (int t = 0; t < T_; ++t) {
    if (d < F_) xs[d] = x[(((size_t)b*T_ + t)*N_ + n)*F_ + d];
    __syncthreads();
    // GRU1
    float gr = g1_bih[d], gz = g1_bih[64+d], gn = g1_bih[128+d];
    for (int f = 0; f < F_; ++f) {
      float xv = xs[f];
      gr += xv * g1_Wih[d*F_+f];
      gz += xv * g1_Wih[(64+d)*F_+f];
      gn += xv * g1_Wih[(128+d)*F_+f];
    }
    float hr = g1_bhh[d], hz = g1_bhh[64+d], hn = g1_bhh[128+d];
    for (int e = 0; e < D_; ++e) {
      float hv = h1[e];
      hr += hv * g1_Whh[d*D_+e];
      hz += hv * g1_Whh[(64+d)*D_+e];
      hn += hv * g1_Whh[(128+d)*D_+e];
    }
    float r  = sigmoidf_(gr + hr);
    float u  = sigmoidf_(gz + hz);
    float nn = tanhf(gn + r*hn);
    float h1n = (1.f-u)*nn + u*h1[d];
    __syncthreads();
    h1[d] = h1n; s1[d] = h1n;
    __syncthreads();
    // GRU2
    float gr2 = g2_bih[d], gz2 = g2_bih[64+d], gn2 = g2_bih[128+d];
    for (int e = 0; e < D_; ++e) {
      float xv = s1[e];
      gr2 += xv * g2_Wih[d*D_+e];
      gz2 += xv * g2_Wih[(64+d)*D_+e];
      gn2 += xv * g2_Wih[(128+d)*D_+e];
    }
    float hr2 = g2_bhh[d], hz2 = g2_bhh[64+d], hn2 = g2_bhh[128+d];
    for (int e = 0; e < D_; ++e) {
      float hv = h2[e];
      hr2 += hv * g2_Whh[d*D_+e];
      hz2 += hv * g2_Whh[(64+d)*D_+e];
      hn2 += hv * g2_Whh[(128+d)*D_+e];
    }
    float r2  = sigmoidf_(gr2 + hr2);
    float u2  = sigmoidf_(gz2 + hz2);
    float nn2 = tanhf(gn2 + r2*hn2);
    float h2n = (1.f-u2)*nn2 + u2*h2[d];
    __syncthreads();
    h2[d] = h2n;
    __syncthreads();
  }
  h_local[(size_t)s*D_ + d] = h2[d];
}

// ---------------------------------------------------------------- LN1 + QKV
__global__ __launch_bounds__(64)
void ln_qkv_kernel(const float* __restrict__ h_local,
                   const float* __restrict__ ln1_g, const float* __restrict__ ln1_b,
                   const float* __restrict__ ain_w, const float* __restrict__ ain_b,
                   float* __restrict__ qkv)
{
  int s = blockIdx.x; int d = threadIdx.x;
  __shared__ float h[D_], a[D_];
  h[d] = h_local[(size_t)s*D_ + d];
  __syncthreads();
  float m = 0.f; for (int e=0;e<D_;++e) m += h[e]; m *= (1.f/D_);
  float v = 0.f; for (int e=0;e<D_;++e){ float t=h[e]-m; v += t*t; } v *= (1.f/D_);
  float inv = rsqrtf(v + 1e-5f);
  a[d] = (h[d]-m)*inv*ln1_g[d] + ln1_b[d];
  __syncthreads();
  float o0 = ain_b[d], o1 = ain_b[64+d], o2 = ain_b[128+d];
  for (int e=0;e<D_;++e){
    float av = a[e];
    o0 += av*ain_w[d*D_+e];
    o1 += av*ain_w[(64+d)*D_+e];
    o2 += av*ain_w[(128+d)*D_+e];
  }
  qkv[(size_t)s*192 + d]       = o0;
  qkv[(size_t)s*192 + 64 + d]  = o1;
  qkv[(size_t)s*192 + 128 + d] = o2;
}

// ---------------------------------------------------------------- attention (per b,head)
__global__ __launch_bounds__(128)
void attn_kernel(const float* __restrict__ qkv, float* __restrict__ o_buf)
{
  int b = blockIdx.x >> 2; int h = blockIdx.x & 3;
  int i = threadIdx.x;
  __shared__ float kh[N_][16], vh[N_][16];
  const float* base = qkv + (size_t)(b*N_ + i)*192;
  for (int d=0; d<16; ++d){ kh[i][d] = base[64 + h*16 + d]; vh[i][d] = base[128 + h*16 + d]; }
  __syncthreads();
  float q[16];
  for (int d=0; d<16; ++d) q[d] = base[h*16 + d];
  float mx = -1e30f;
  for (int j=0;j<N_;++j){
    float sc = 0.f;
    for (int d=0;d<16;++d) sc += q[d]*kh[j][d];
    mx = fmaxf(mx, sc*0.25f);
  }
  float sum = 0.f, acc[16];
  for (int d=0;d<16;++d) acc[d]=0.f;
  for (int j=0;j<N_;++j){
    float sc = 0.f;
    for (int d=0;d<16;++d) sc += q[d]*kh[j][d];
    float p = __expf(sc*0.25f - mx);
    sum += p;
    for (int d=0;d<16;++d) acc[d] += p*vh[j][d];
  }
  float rs = 1.f/sum;
  float* op = o_buf + (size_t)(b*N_ + i)*D_ + h*16;
  for (int d=0;d<16;++d) op[d] = acc[d]*rs;
}

// ---------------------------------------------------------------- residual + LN2 + FFN -> emb
__global__ __launch_bounds__(64)
void fuse_ffn_kernel(const float* __restrict__ h_local, const float* __restrict__ o_buf,
                     const float* __restrict__ aout_w, const float* __restrict__ aout_b,
                     const float* __restrict__ ln2_g, const float* __restrict__ ln2_b,
                     const float* __restrict__ ff1_w, const float* __restrict__ ff1_b,
                     const float* __restrict__ ff2_w, const float* __restrict__ ff2_b,
                     float* __restrict__ emb)
{
  int s = blockIdx.x; int d = threadIdx.x;
  __shared__ float ob[D_], hf[D_], fi[D_], hid[256];
  ob[d] = o_buf[(size_t)s*D_+d];
  __syncthreads();
  float acc = aout_b[d];
  for (int e=0;e<D_;++e) acc += ob[e]*aout_w[d*D_+e];
  float hfd = h_local[(size_t)s*D_+d] + acc;
  hf[d] = hfd;
  __syncthreads();
  float m=0.f; for (int e=0;e<D_;++e) m += hf[e]; m *= (1.f/D_);
  float v=0.f; for (int e=0;e<D_;++e){ float t=hf[e]-m; v += t*t; } v *= (1.f/D_);
  float inv = rsqrtf(v+1e-5f);
  fi[d] = (hfd-m)*inv*ln2_g[d] + ln2_b[d];
  __syncthreads();
  for (int r=0;r<4;++r){
    int u = d + r*64;
    float a = ff1_b[u];
    for (int e=0;e<D_;++e) a += fi[e]*ff1_w[u*D_+e];
    hid[u] = geluf_(a);
  }
  __syncthreads();
  float e2 = ff2_b[d];
  for (int u=0;u<256;++u) e2 += hid[u]*ff2_w[d*256+u];
  emb[(size_t)s*D_+d] = hfd + e2;
}

// ---------------------------------------------------------------- ga GEMV: gi = emb_flat @ Wih.T + b
__global__ __launch_bounds__(256)
void ga_gemv_kernel(const float* __restrict__ emb, const float* __restrict__ Wih,
                    const float* __restrict__ bih, float* __restrict__ gi)
{
  int wv = (blockIdx.x*blockDim.x + threadIdx.x) >> 5;  // output unit u (0..383)
  int lane = threadIdx.x & 31;
  if (wv >= 384) return;
  const float* wr = Wih + (size_t)wv*8192;
  float a0=0.f, a1=0.f, a2=0.f, a3=0.f;
  for (int k=lane; k<8192; k+=32){
    float w = wr[k];
    a0 += emb[k]*w;
    a1 += emb[8192+k]*w;
    a2 += emb[16384+k]*w;
    a3 += emb[24576+k]*w;
  }
  for (int m=16;m>=1;m>>=1){
    a0 += __shfl_xor(a0, m, 32);
    a1 += __shfl_xor(a1, m, 32);
    a2 += __shfl_xor(a2, m, 32);
    a3 += __shfl_xor(a3, m, 32);
  }
  if (lane==0){
    float bv = bih[wv];
    gi[wv] = a0+bv; gi[384+wv] = a1+bv; gi[768+wv] = a2+bv; gi[1152+wv] = a3+bv;
  }
}

// ---------------------------------------------------------------- latent + flows (per batch)
__global__ __launch_bounds__(256)
void latent_kernel(const float* __restrict__ gi, const float* __restrict__ ga_bhh,
                   const float* __restrict__ s2l_w, const float* __restrict__ s2l_b,
                   const float* __restrict__ fd1_w, const float* __restrict__ fd1_b,
                   const float* __restrict__ fd2_w, const float* __restrict__ fd2_b,
                   const float* __restrict__ fi1_w, const float* __restrict__ fi1_b,
                   const float* __restrict__ fi2_w, const float* __restrict__ fi2_b,
                   const float* __restrict__ fe1_w, const float* __restrict__ fe1_b,
                   const float* __restrict__ fe2_w, const float* __restrict__ fe2_b,
                   const float* __restrict__ e2i_w, const float* __restrict__ i2e_w,
                   float* __restrict__ zd_out, float* __restrict__ zi_out, float* __restrict__ ze_out)
{
  int b = blockIdx.x; int t = threadIdx.x;
  __shared__ float s[GS_], z[GS_], td[130], ti[66], te[66];
  const float* g = gi + b*384;
  if (t < 128){
    float r  = sigmoidf_(g[t] + ga_bhh[t]);
    float u  = sigmoidf_(g[128+t] + ga_bhh[128+t]);
    float nn = tanhf(g[256+t] + r*ga_bhh[256+t]);
    s[t] = (1.f-u)*nn;
  }
  __syncthreads();
  if (t < 128){
    float a = s2l_b[t];
    for (int k=0;k<128;++k) a += s[k]*s2l_w[t*128+k];
    z[t] = a;  // mu
  }
  __syncthreads();
  if (t < 130){
    float a = fd1_b[t];
    for (int k=0;k<64;++k) a += z[k]*fd1_w[t*65+k];   // t0 element is zero
    td[t] = tanhf(a);
  }
  if (t < 66){
    float a = fi1_b[t];
    for (int k=0;k<32;++k) a += z[64+k]*fi1_w[t*33+k];
    ti[t] = tanhf(a);
    float c = fe1_b[t];
    for (int k=0;k<32;++k) c += z[96+k]*fe1_w[t*33+k];
    te[t] = tanhf(c);
  }
  __syncthreads();
  if (t < 64){
    float a = fd2_b[t];
    for (int u=0;u<130;++u) a += td[u]*fd2_w[t*130+u];
    zd_out[b*64+t] = z[t] + a;
  }
  if (t < 32){
    float a = fi2_b[t];
    for (int u=0;u<66;++u) a += ti[u]*fi2_w[t*66+u];
    float cx = 0.f;
    for (int k=0;k<32;++k) cx += z[96+k]*e2i_w[t*32+k];
    zi_out[b*32+t] = z[64+t] + a + 0.1f*cx;

    float c = fe2_b[t];
    for (int u=0;u<66;++u) c += te[u]*fe2_w[t*66+u];
    float cx2 = 0.f;
    for (int k=0;k<32;++k) cx2 += z[64+k]*i2e_w[t*32+k];
    ze_out[b*32+t] = z[96+t] + c + 0.1f*cx2;
  }
}

// ---------------------------------------------------------------- per-node projections for pairwise MLPs
__global__ __launch_bounds__(128)
void pair_proj_kernel(const float* __restrict__ emb,
                      const float* __restrict__ iw1, const float* __restrict__ lw1,
                      float* __restrict__ PiI, float* __restrict__ PjI,
                      float* __restrict__ PiL, float* __restrict__ PjL)
{
  int s = blockIdx.x; int u = threadIdx.x;
  __shared__ float e[D_];
  if (u < D_) e[u] = emb[(size_t)s*D_+u];
  __syncthreads();
  float a=0.f, bm=0.f, c=0.f, dm=0.f;
  for (int k=0;k<D_;++k){
    float ev = e[k];
    a  += ev*iw1[u*160+k];
    bm += ev*iw1[u*160+64+k];
    c  += ev*lw1[u*168+k];
    dm += ev*lw1[u*168+64+k];
  }
  PiI[(size_t)s*128+u]=a;  PjI[(size_t)s*128+u]=bm;
  PiL[(size_t)s*128+u]=c;  PjL[(size_t)s*128+u]=dm;
}

__global__ __launch_bounds__(128)
void pair_base_kernel(const float* __restrict__ zi, const float* __restrict__ ze,
                      const float* __restrict__ lag_emb,
                      const float* __restrict__ iw1, const float* __restrict__ ib1,
                      const float* __restrict__ lw1, const float* __restrict__ lb1,
                      float* __restrict__ baseI, float* __restrict__ baseL)
{
  int b = blockIdx.x; int u = threadIdx.x;
  float a = ib1[u];
  for (int k=0;k<32;++k) a += zi[b*32+k]*iw1[u*160+128+k];
  baseI[b*128+u] = a;
  float zp = lb1[u];
  for (int k=0;k<32;++k) zp += ze[b*32+k]*lw1[u*168+128+k];
  for (int l=0;l<L_;++l){
    float c = zp;
    for (int k=0;k<8;++k) c += lag_emb[l*8+k]*lw1[u*168+160+k];
    baseL[(b*L_+l)*128+u] = c;
  }
}

// ---------------------------------------------------------------- f32 -> f16
__global__ __launch_bounds__(256)
void f2h_kernel(const float* __restrict__ src, _Float16* __restrict__ dst, int n)
{
  int i = blockIdx.x*blockDim.x + threadIdx.x;
  if (i < n) dst[i] = (_Float16)src[i];
}

// ---------------------------------------------------------------- pairwise MLP via WMMA
// h1 = relu(Pi[b,i]+Pj[b,j]+base[g]);  h2 = relu(h1 @ W2.T + b2);  out = sigmoid(h2.w3 + b3)*mask
__global__ __launch_bounds__(256)
void pair_mlp_wmma_kernel(const float* __restrict__ Pi, const float* __restrict__ Pj,
                          const float* __restrict__ base,
                          const _Float16* __restrict__ W2h, const float* __restrict__ b2,
                          const float* __restrict__ w3, const float* __restrict__ b3,
                          float* __restrict__ out, int Ldiv)
{
  int wave = (blockIdx.x * blockDim.x + threadIdx.x) >> 5;
  int lane = threadIdx.x & 31;
  int jt = wave & 7;            // 8 j-tiles of 16
  int i  = (wave >> 3) & (N_-1);
  int g  = wave >> 10;          // batch*lag group
  int b  = g / Ldiv;
  int j0 = jt * 16;
  int r  = lane & 15;
  int hi = lane >> 4;
  const float* pi = Pi + ((size_t)b*N_ + i)*128;
  const float* pj = Pj + ((size_t)b*N_ + j0 + r)*128;
  const float* bs = base + (size_t)g*128;

  v8f acc0 = {0.f,0.f,0.f,0.f,0.f,0.f,0.f,0.f};
  v8f acc1 = acc0, acc2 = acc0, acc3 = acc0;

  #pragma unroll
  for (int kc = 0; kc < 4; ++kc) {
    // A fragment (16x32 f16): lane r holds row M=r; lanes<16 own K {0..15}, lanes>=16 own {16..31}
    v16h a;
    int kb = kc*32 + hi*16;
    #pragma unroll
    for (int t = 0; t < 8; ++t) {
      float v0 = pi[kb+t]    + pj[kb+t]    + bs[kb+t];
      float v1 = pi[kb+8+t]  + pj[kb+8+t]  + bs[kb+8+t];
      a[t]   = (_Float16)fmaxf(v0, 0.0f);
      a[8+t] = (_Float16)fmaxf(v1, 0.0f);
    }
    // B fragment (32x16 f16): lane holds column n = nt*16 + r, 16 consecutive K == row of W2 (row-major)
    const _Float16* wb = W2h + (size_t)r*128 + kc*32 + hi*16;
    v16h bf0 = *(const v16h*)(wb);
    v16h bf1 = *(const v16h*)(wb + 16*128);
    v16h bf2 = *(const v16h*)(wb + 32*128);
    v16h bf3 = *(const v16h*)(wb + 48*128);
    acc0 = __builtin_amdgcn_wmma_f32_16x16x32_f16(false, a, false, bf0, (short)0, acc0, false, false);
    acc1 = __builtin_amdgcn_wmma_f32_16x16x32_f16(false, a, false, bf1, (short)0, acc1, false, false);
    acc2 = __builtin_amdgcn_wmma_f32_16x16x32_f16(false, a, false, bf2, (short)0, acc2, false, false);
    acc3 = __builtin_amdgcn_wmma_f32_16x16x32_f16(false, a, false, bf3, (short)0, acc3, false, false);
  }
  // epilogue: relu(+b2) dot w3, reduced across the 16 columns each half-wave owns
  float b2v0 = b2[r],    b2v1 = b2[16+r], b2v2 = b2[32+r], b2v3 = b2[48+r];
  float w3v0 = w3[r],    w3v1 = w3[16+r], w3v2 = w3[32+r], w3v3 = w3[48+r];
  float part[8];
  #pragma unroll
  for (int v = 0; v < 8; ++v) {
    part[v] = fmaxf(acc0[v] + b2v0, 0.f)*w3v0
            + fmaxf(acc1[v] + b2v1, 0.f)*w3v1
            + fmaxf(acc2[v] + b2v2, 0.f)*w3v2
            + fmaxf(acc3[v] + b2v3, 0.f)*w3v3;
  }
  #pragma unroll
  for (int m = 8; m >= 1; m >>= 1) {
    #pragma unroll
    for (int v = 0; v < 8; ++v) part[v] += __shfl_xor(part[v], m, 32);
  }
  if (r == 0) {
    float bb = b3[0];
    #pragma unroll
    for (int v = 0; v < 8; ++v) {
      int j = j0 + hi*8 + v;  // C layout: vgpr v of lane half hi is row M = v + hi*8
      float wl = part[v] + bb;
      out[((size_t)g*N_ + i)*N_ + j] = (i == j) ? 0.0f : sigmoidf_(wl);
    }
  }
}

// ---------------------------------------------------------------- gate/trans -> cur, cur_p
__global__ __launch_bounds__(64)
void gate_cur_kernel(const float* __restrict__ emb, const float* __restrict__ zd,
                     const float* __restrict__ fg_w, const float* __restrict__ fg_b,
                     const float* __restrict__ ft_w, const float* __restrict__ ft_b,
                     const float* __restrict__ ip_w, const float* __restrict__ ip_b,
                     float* __restrict__ cur, float* __restrict__ cur_p)
{
  int s = blockIdx.x; int b = s >> 7; int d = threadIdx.x;
  __shared__ float e[D_], zds[DD_], c[D_];
  e[d]   = emb[(size_t)s*D_+d];
  zds[d] = zd[b*64+d];
  __syncthreads();
  float ag = fg_b[d], at = ft_b[d];
  for (int k=0;k<64;++k){
    ag += e[k]*fg_w[d*128+k];
    at += e[k]*ft_w[d*128+k];
  }
  for (int k=0;k<64;++k){
    ag += zds[k]*fg_w[d*128+64+k];
    at += zds[k]*ft_w[d*128+64+k];
  }
  float gt = sigmoidf_(ag);
  float tr = fmaxf(at, 0.f);
  float cv = (1.f-gt)*e[d] + gt*tr;
  c[d] = cv;
  cur[(size_t)s*D_+d] = cv;
  __syncthreads();
  float a = ip_b[d];
  for (int k=0;k<64;++k) a += c[k]*ip_w[d*64+k];
  cur_p[(size_t)s*D_+d] = a;
}

// ---------------------------------------------------------------- lag messages
__global__ __launch_bounds__(64)
void lag_msg_kernel(const float* __restrict__ x, const float* __restrict__ lp_w,
                    const float* __restrict__ lp_b, float* __restrict__ lag_msg)
{
  int blk = blockIdx.x;              // b*L*N + l*N + n
  int n = blk & 127;
  int l = (blk >> 7) % L_;
  int b = blk / (L_*N_);
  int d = threadIdx.x;
  const float* xp = x + (((size_t)b*T_ + (T_-L_+l))*N_ + n)*F_;
  float a = lp_b[d];
  for (int f=0; f<F_; ++f) a += xp[f]*lp_w[d*F_+f];
  lag_msg[(size_t)blk*D_ + d] = a;
}

// ---------------------------------------------------------------- aggregation + decoder -> x_pred
__global__ __launch_bounds__(64)
void final_kernel(const float* __restrict__ W_t, const float* __restrict__ A_lags,
                  const float* __restrict__ cur, const float* __restrict__ cur_p,
                  const float* __restrict__ lag_msg,
                  const float* __restrict__ d1_w, const float* __restrict__ d1_b,
                  const float* __restrict__ d2_w, const float* __restrict__ d2_b,
                  float* __restrict__ x_pred)
{
  int s = blockIdx.x; int b = s >> 7; int i = s & 127; int d = threadIdx.x;
  __shared__ float wrow[N_], arow[N_], rep[D_], hid[128];
  wrow[d]    = W_t[((size_t)b*N_+i)*N_ + d];
  wrow[64+d] = W_t[((size_t)b*N_+i)*N_ + 64 + d];
  __syncthreads();
  float acc = cur[(size_t)s*D_+d];
  for (int j=0;j<N_;++j) acc += wrow[j]*cur_p[((size_t)b*N_+j)*D_+d];
  for (int l=0;l<L_;++l){
    __syncthreads();
    arow[d]    = A_lags[(((size_t)(b*L_+l))*N_+i)*N_ + d];
    arow[64+d] = A_lags[(((size_t)(b*L_+l))*N_+i)*N_ + 64 + d];
    __syncthreads();
    const float* lm = lag_msg + (size_t)(b*L_+l)*N_*D_;
    for (int k=0;k<N_;++k) acc += arow[k]*lm[k*D_+d];
  }
  rep[d] = acc;
  __syncthreads();
  for (int rr=0; rr<2; ++rr){
    int u = d + rr*64;
    float a = d1_b[u];
    for (int e=0;e<D_;++e) a += rep[e]*d1_w[u*64+e];
    hid[u] = fmaxf(a, 0.f);
  }
  __syncthreads();
  if (d < F_){
    float a = d2_b[d];
    for (int u=0;u<128;++u) a += hid[u]*d2_w[d*128+u];
    x_pred[(size_t)s*F_ + d] = a;
  }
}

// ---------------------------------------------------------------- DAG penalty
__global__ __launch_bounds__(256)
void dag_kernel(const float* __restrict__ W_t, float* __restrict__ dagp)
{
  int b = blockIdx.x; int t = threadIdx.x;
  const float* W = W_t + (size_t)b*N_*N_;
  float tr2 = 0.f, tr3 = 0.f, trw = 0.f;
  for (int p = t; p < N_*N_; p += 256){
    int i = p >> 7, j = p & 127;
    float wij = W[i*N_+j];
    float wji = W[j*N_+i];
    tr2 += wij*wji;
    float a2 = 0.f;
    for (int k=0;k<N_;++k) a2 += W[i*N_+k]*W[k*N_+j];
    tr3 += a2*wji;
    if (i==j) trw += wij;
  }
  __shared__ float r2[256], r3[256], rw[256];
  r2[t]=tr2; r3[t]=tr3; rw[t]=trw;
  __syncthreads();
  for (int st=128; st>=1; st>>=1){
    if (t < st){ r2[t]+=r2[t+st]; r3[t]+=r3[t+st]; rw[t]+=rw[t+st]; }
    __syncthreads();
  }
  if (t==0){ dagp[b*3+0]=rw[0]; dagp[b*3+1]=r2[0]; dagp[b*3+2]=r3[0]; }
}

__global__ void dag_fin_kernel(const float* __restrict__ dagp, float* __restrict__ out)
{
  if (threadIdx.x==0 && blockIdx.x==0){
    float trw=0.f, t2=0.f, t3=0.f;
    for (int b=0;b<B_;++b){ trw += fabsf(dagp[b*3+0]); t2 += dagp[b*3+1]; t3 += dagp[b*3+2]; }
    out[0] = 0.5f*trw + 0.1f*(fabsf(t2)+fabsf(t3));
  }
}

// ================================================================ launcher
extern "C" void kernel_launch(void* const* d_in, const int* in_sizes, int n_in,
                              void* d_out, int out_size, void* d_ws, size_t ws_size,
                              hipStream_t stream)
{
  (void)in_sizes; (void)n_in; (void)out_size; (void)ws_size;
  const float* x       = (const float*)d_in[0];
  const float* g1_Wih  = (const float*)d_in[1];
  const float* g1_bih  = (const float*)d_in[2];
  const float* g1_Whh  = (const float*)d_in[3];
  const float* g1_bhh  = (const float*)d_in[4];
  const float* g2_Wih  = (const float*)d_in[5];
  const float* g2_bih  = (const float*)d_in[6];
  const float* g2_Whh  = (const float*)d_in[7];
  const float* g2_bhh  = (const float*)d_in[8];
  const float* ain_w   = (const float*)d_in[9];
  const float* ain_b   = (const float*)d_in[10];
  const float* aout_w  = (const float*)d_in[11];
  const float* aout_b  = (const float*)d_in[12];
  const float* ln1_g   = (const float*)d_in[13];
  const float* ln1_b   = (const float*)d_in[14];
  const float* ln2_g   = (const float*)d_in[15];
  const float* ln2_b   = (const float*)d_in[16];
  const float* ff1_w   = (const float*)d_in[17];
  const float* ff1_b   = (const float*)d_in[18];
  const float* ff2_w   = (const float*)d_in[19];
  const float* ff2_b   = (const float*)d_in[20];
  const float* ga_Wih  = (const float*)d_in[21];
  const float* ga_bih  = (const float*)d_in[22];
  /* ga_Whh (d_in[23]) unused: h0 == 0 */
  const float* ga_bhh  = (const float*)d_in[24];
  const float* s2l_w   = (const float*)d_in[25];
  const float* s2l_b   = (const float*)d_in[26];
  const float* fd1_w   = (const float*)d_in[27];
  const float* fd1_b   = (const float*)d_in[28];
  const float* fd2_w   = (const float*)d_in[29];
  const float* fd2_b   = (const float*)d_in[30];
  const float* fi1_w   = (const float*)d_in[31];
  const float* fi1_b   = (const float*)d_in[32];
  const float* fi2_w   = (const float*)d_in[33];
  const float* fi2_b   = (const float*)d_in[34];
  const float* fe1_w   = (const float*)d_in[35];
  const float* fe1_b   = (const float*)d_in[36];
  const float* fe2_w   = (const float*)d_in[37];
  const float* fe2_b   = (const float*)d_in[38];
  const float* e2i_w   = (const float*)d_in[39];
  const float* i2e_w   = (const float*)d_in[40];
  const float* lag_emb = (const float*)d_in[41];
  const float* iw1     = (const float*)d_in[42];
  const float* ib1     = (const float*)d_in[43];
  const float* iw2     = (const float*)d_in[44];
  const float* ib2     = (const float*)d_in[45];
  const float* iw3     = (const float*)d_in[46];
  const float* ib3     = (const float*)d_in[47];
  const float* lw1     = (const float*)d_in[48];
  const float* lb1     = (const float*)d_in[49];
  const float* lw2     = (const float*)d_in[50];
  const float* lb2     = (const float*)d_in[51];
  const float* lw3     = (const float*)d_in[52];
  const float* lb3     = (const float*)d_in[53];
  const float* fg_w    = (const float*)d_in[54];
  const float* fg_b    = (const float*)d_in[55];
  const float* ft_w    = (const float*)d_in[56];
  const float* ft_b    = (const float*)d_in[57];
  const float* ip_w    = (const float*)d_in[58];
  const float* ip_b    = (const float*)d_in[59];
  const float* lp_w    = (const float*)d_in[60];
  const float* lp_b    = (const float*)d_in[61];
  const float* d1_w    = (const float*)d_in[62];
  const float* d1_b    = (const float*)d_in[63];
  const float* d2_w    = (const float*)d_in[64];
  const float* d2_b    = (const float*)d_in[65];

  float* ws = (float*)d_ws;
  size_t off = 0;
  float* h_local = ws + off; off += 32768;           // (B,N,64)
  float* qkv     = ws + off; off += 98304;           // (B,N,192)
  float* o_buf   = ws + off; off += 32768;           // (B,N,64)
  float* emb     = ws + off; off += 32768;           // (B,N,64)
  float* gi      = ws + off; off += 1536;            // (B,384)
  float* zd      = ws + off; off += 256;             // (B,64)
  float* zi      = ws + off; off += 128;             // (B,32)
  float* ze      = ws + off; off += 128;             // (B,32)
  float* PiI     = ws + off; off += 65536;           // (B,N,128)
  float* PjI     = ws + off; off += 65536;
  float* PiL     = ws + off; off += 65536;
  float* PjL     = ws + off; off += 65536;
  float* baseI   = ws + off; off += 512;             // (B,128)
  float* baseL   = ws + off; off += 2560;            // (B*L,128)
  float* cur     = ws + off; off += 32768;           // (B,N,64)
  float* cur_p   = ws + off; off += 32768;           // (B,N,64)
  float* lagm    = ws + off; off += 163840;          // (B,L,N,64)
  float* dagp    = ws + off; off += 16;
  off = (off + 15) & ~(size_t)15;
  _Float16* iw2_h = (_Float16*)(ws + off); off += 4096;  // 8192 halfs
  _Float16* lw2_h = (_Float16*)(ws + off); off += 4096;

  float* out     = (float*)d_out;
  float* o_xpred = out;                 // (B,N,F)    8192
  float* o_Wt    = out + 8192;          // (B,N,N)    65536
  float* o_Alag  = out + 73728;         // (B,L,N,N)  327680
  float* o_dag   = out + 401408;        // scalar

  // weight conversion for WMMA B fragments
  f2h_kernel<<<32, 256, 0, stream>>>(iw2, iw2_h, 8192);
  f2h_kernel<<<32, 256, 0, stream>>>(lw2, lw2_h, 8192);

  // encoder
  gru_kernel<<<512, 64, 0, stream>>>(x, g1_Wih, g1_bih, g1_Whh, g1_bhh,
                                     g2_Wih, g2_bih, g2_Whh, g2_bhh, h_local);
  ln_qkv_kernel<<<512, 64, 0, stream>>>(h_local, ln1_g, ln1_b, ain_w, ain_b, qkv);
  attn_kernel<<<B_*NH_, 128, 0, stream>>>(qkv, o_buf);
  fuse_ffn_kernel<<<512, 64, 0, stream>>>(h_local, o_buf, aout_w, aout_b,
                                          ln2_g, ln2_b, ff1_w, ff1_b, ff2_w, ff2_b, emb);

  // latent path
  ga_gemv_kernel<<<48, 256, 0, stream>>>(emb, ga_Wih, ga_bih, gi);
  latent_kernel<<<B_, 256, 0, stream>>>(gi, ga_bhh, s2l_w, s2l_b,
                                        fd1_w, fd1_b, fd2_w, fd2_b,
                                        fi1_w, fi1_b, fi2_w, fi2_b,
                                        fe1_w, fe1_b, fe2_w, fe2_b,
                                        e2i_w, i2e_w, zd, zi, ze);

  // pairwise MLP factorization
  pair_proj_kernel<<<512, 128, 0, stream>>>(emb, iw1, lw1, PiI, PjI, PiL, PjL);
  pair_base_kernel<<<B_, 128, 0, stream>>>(zi, ze, lag_emb, iw1, ib1, lw1, lb1, baseI, baseL);

  // WMMA pairwise MLPs: inst (G=4) -> W_t ; lag (G=20) -> A_lags
  pair_mlp_wmma_kernel<<<512, 256, 0, stream>>>(PiI, PjI, baseI, iw2_h, ib2, iw3, ib3, o_Wt, 1);
  pair_mlp_wmma_kernel<<<2560, 256, 0, stream>>>(PiL, PjL, baseL, lw2_h, lb2, lw3, lb3, o_Alag, L_);

  // decoder path
  gate_cur_kernel<<<512, 64, 0, stream>>>(emb, zd, fg_w, fg_b, ft_w, ft_b, ip_w, ip_b, cur, cur_p);
  lag_msg_kernel<<<B_*L_*N_, 64, 0, stream>>>(x, lp_w, lp_b, lagm);
  final_kernel<<<512, 64, 0, stream>>>(o_Wt, o_Alag, cur, cur_p, lagm,
                                       d1_w, d1_b, d2_w, d2_b, o_xpred);

  // DAG penalty
  dag_kernel<<<B_, 256, 0, stream>>>(o_Wt, dagp);
  dag_fin_kernel<<<1, 32, 0, stream>>>(dagp, o_dag);
}
